// MultiHeadAttention_47382079209593
// MI455X (gfx1250) — matrix operand order repaired and checked by reading, not compiled, not run on hardware
//
#include <hip/hip_runtime.h>

// ---------------------------------------------------------------------------
// MI455X / gfx1250, wave32.  bf16 WMMA (f32 accumulate) everywhere.
// ---------------------------------------------------------------------------

typedef __bf16 bf16_t;
typedef bf16_t v16bf __attribute__((ext_vector_type(16)));
typedef bf16_t v8bf  __attribute__((ext_vector_type(8)));
typedef bf16_t v4bf  __attribute__((ext_vector_type(4)));
typedef bf16_t v2bf  __attribute__((ext_vector_type(2)));
typedef float  v8f   __attribute__((ext_vector_type(8)));
typedef float  v4f   __attribute__((ext_vector_type(4)));

#define WMMA_BF16(a, b, c) \
  __builtin_amdgcn_wmma_f32_16x16x32_bf16(false, (a), false, (b), (short)0, (c), false, false)

__device__ __forceinline__ v16bf pack16(v8bf lo, v8bf hi) {
  return __builtin_shufflevector(lo, hi, 0,1,2,3,4,5,6,7,8,9,10,11,12,13,14,15);
}
__device__ __forceinline__ v2bf cvt2(float a, float b) {          // -> v_cvt_pk_bf16_f32
  v2bf h; h[0] = (bf16_t)a; h[1] = (bf16_t)b; return h;
}
__device__ __forceinline__ v8bf cvt8(v4f a, v4f b) {
  v8bf h;
  #pragma unroll
  for (int i = 0; i < 4; ++i) { h[i] = (bf16_t)a[i]; h[i + 4] = (bf16_t)b[i]; }
  return h;
}

// ---------------------------------------------------------------------------
// GEMM: C[M,N] = A[M,K] * B[K,N], f32 memory, bf16 WMMA compute.
// 256 threads = 8 waves; block tile 128(M) x 128(N); K pipelined 32-deep with
// double-buffered LDS (global loads for tile k+1 issued before the WMMAs of
// tile k; f32->bf16 convert + ds_store after, so HBM latency hides under the
// matrix pipe).  Wave grid 4x2; each wave owns 32x64 = 2x4 WMMA tiles.
// A-row accesses are clamped (not guarded) to M-1: out-of-range rows only
// produce C rows >= M which are never stored -> unconditional batched loads.
// B transpose staging: each thread owns 2 k-rows x 8 n-cols so each (k,k+1)
// pair packs via one v_cvt_pk_bf16_f32 into a single ds_store_b32.
// ---------------------------------------------------------------------------
__global__ __launch_bounds__(256)
void gemm_bf16_wmma(const float* __restrict__ A, const float* __restrict__ B,
                    float* __restrict__ C, int M, int N, int K,
                    int lda, int ldb, int ldc)
{
  __shared__ __align__(16) bf16_t As[2][128][40];   // [buf][m][k], 80B rows
  __shared__ __align__(16) bf16_t Bs[2][128][40];   // [buf][n][k]  (transposed)

  const int tid    = threadIdx.x;
  const int lane   = tid & 31;
  const int wave   = tid >> 5;          // 0..7
  const int wm     = (wave >> 1) * 32;  // 0,32,64,96
  const int wn     = (wave & 1) * 64;   // 0,64
  const int half16 = (lane & 16) ? 1 : 0;
  const int l15    = lane & 15;

  const int m0 = blockIdx.y * 128;
  const int n0 = blockIdx.x * 128;

  // Per-thread staging coordinates (constant across K iterations).
  const int ar  = tid >> 1;             // A tile row   0..127
  const int ac  = (tid & 1) * 16;       // A tile col base (16 f32 / thread)
  const int bk2 = (tid >> 4) * 2;       // B tile k-row pair 0,2,..,30
  const int bn8 = (tid & 15) * 8;       // B tile n base (8 f32 x 2 rows / thread)

  const int arow_c = (m0 + ar < M) ? (m0 + ar) : (M - 1);   // clamp, no guard
  const float* Aptr = A + (size_t)arow_c * lda + ac;
  const float* Bptr = B + (size_t)bk2 * ldb + n0 + bn8;

  v4f aR[4], bR[2][2];
  auto load_tiles = [&](int k0) {
    #pragma unroll
    for (int i = 0; i < 4; ++i) aR[i] = *(const v4f*)(Aptr + k0 + i * 4);
    #pragma unroll
    for (int rr = 0; rr < 2; ++rr)
      #pragma unroll
      for (int i = 0; i < 2; ++i)
        bR[rr][i] = *(const v4f*)(Bptr + (size_t)(k0 + rr) * ldb + i * 4);
  };
  auto store_tiles = [&](int buf) {
    *(v8bf*)&As[buf][ar][ac]     = cvt8(aR[0], aR[1]);
    *(v8bf*)&As[buf][ar][ac + 8] = cvt8(aR[2], aR[3]);
    #pragma unroll
    for (int n = 0; n < 8; ++n)          // pack (k,k+1) -> one b32 store
      *(v2bf*)&Bs[buf][bn8 + n][bk2] =
          cvt2(bR[0][n >> 2][n & 3], bR[1][n >> 2][n & 3]);
  };

  const v8f zero8 = {0.f,0.f,0.f,0.f,0.f,0.f,0.f,0.f};
  v8f acc[2][4];
  #pragma unroll
  for (int s = 0; s < 2; ++s)
    #pragma unroll
    for (int j = 0; j < 4; ++j) acc[s][j] = zero8;

  load_tiles(0);
  store_tiles(0);
  __syncthreads();

  int cur = 0;
  for (int k0 = 0; k0 < K; k0 += 32) {
    const bool more = (k0 + 32 < K);
    if (more) load_tiles(k0 + 32);            // issue early: overlap with WMMA
    if (k0 + 64 < K) {
      __builtin_prefetch(Aptr + k0 + 64, 0, 3);
      __builtin_prefetch(Bptr + (size_t)(k0 + 64) * ldb, 0, 3);
    }

    // ---- fragments + 8 WMMA from buffer `cur` ----
    v16bf afrag[2], bfrag[4];
    #pragma unroll
    for (int s = 0; s < 2; ++s) {
      const int r  = wm + s * 16 + l15;
      const int c0 = half16 ? 8 : 0;          // A: K chunks {c0..c0+7},{c0+16..}
      afrag[s] = pack16(*(const v8bf*)&As[cur][r][c0],
                        *(const v8bf*)&As[cur][r][c0 + 16]);
    }
    #pragma unroll
    for (int j = 0; j < 4; ++j) {
      const int n  = wn + j * 16 + l15;
      const int kk = half16 ? 8 : 0;          // B: K chunks {kk..kk+7},{kk+16..}
      bfrag[j] = pack16(*(const v8bf*)&Bs[cur][n][kk],
                        *(const v8bf*)&Bs[cur][n][kk + 16]);
    }
    #pragma unroll
    for (int s = 0; s < 2; ++s)
      #pragma unroll
      for (int j = 0; j < 4; ++j)
        acc[s][j] = WMMA_BF16(afrag[s], bfrag[j], acc[s][j]);

    if (more) store_tiles(cur ^ 1);           // waits for loads after WMMA issue
    __syncthreads();
    cur ^= 1;
  }

  // ---- epilogue: D layout VGPR r -> row r (lanes 0-15) / r+8 (lanes 16-31)
  #pragma unroll
  for (int s = 0; s < 2; ++s)
    #pragma unroll
    for (int j = 0; j < 4; ++j)
      #pragma unroll
      for (int r = 0; r < 8; ++r) {
        const int row = m0 + wm + s * 16 + r + half16 * 8;
        const int col = n0 + wn + j * 16 + l15;
        if (row < M)
          C[(size_t)row * ldc + col] = acc[s][j][r];
      }
}

// ---------------------------------------------------------------------------
// Causal flash attention over qkv = [B=64][W=512][3*512] f32.
// Block = 128 threads (4 waves) handles one (b, h, 64 q-rows); each wave
// owns 16 q rows.  K/V staged 32 keys at a time in LDS; QK^T and P*V via
// bf16 WMMA; online softmax in f32 using 16-lane xor-shuffle reductions.
// ---------------------------------------------------------------------------
__global__ __launch_bounds__(128)
void attention_kernel(const float* __restrict__ qkv, float* __restrict__ out)
{
  const int qblk = blockIdx.x;          // 0..7 (q tile of 64)
  const int h    = blockIdx.y;          // 0..7
  const int b    = blockIdx.z;          // 0..63
  const int tid  = threadIdx.x;
  const int lane = tid & 31;
  const int wave = tid >> 5;            // 0..3
  const int half16 = (lane & 16) ? 1 : 0;
  const int l15  = lane & 15;
  const int qb   = qblk * 64;
  const int qr0  = qb + wave * 16;

  __shared__ __align__(16) bf16_t Ks[32][72];     // [k][d]
  __shared__ __align__(16) bf16_t Vs[64][40];     // [d][k]  (transposed)
  __shared__ __align__(16) bf16_t Ps[4][16][40];  // per-wave P (16q x 32k)

  const size_t rstride = 1536;  // 3*C

  // ---- preload Q fragments (kept in registers across the k loop) ----
  const float* qrow = qkv + ((size_t)b * 512 + (qr0 + l15)) * rstride + h * 64;
  v16bf qfrag[2];
  #pragma unroll
  for (int s = 0; s < 2; ++s) {
    const int c0 = s * 32 + (half16 ? 8 : 0);
    v4f f0 = *(const v4f*)(qrow + c0);
    v4f f1 = *(const v4f*)(qrow + c0 + 4);
    v4f g0 = *(const v4f*)(qrow + c0 + 16);
    v4f g1 = *(const v4f*)(qrow + c0 + 20);
    v16bf a;
    #pragma unroll
    for (int i = 0; i < 4; ++i) {
      a[i]      = (bf16_t)f0[i];
      a[i + 4]  = (bf16_t)f1[i];
      a[i + 8]  = (bf16_t)g0[i];
      a[i + 12] = (bf16_t)g1[i];
    }
    qfrag[s] = a;
  }

  // Staging coords: each thread owns 2 key rows x 8 d columns.
  const int skr2 = (tid >> 3) * 2;      // key row pair 0,2,..,30
  const int sd8  = (tid & 7) * 8;       // d base

  const v8f zero8 = {0.f,0.f,0.f,0.f,0.f,0.f,0.f,0.f};
  v8f o[4] = {zero8, zero8, zero8, zero8};
  float mrow[8], lrow[8];
  #pragma unroll
  for (int r = 0; r < 8; ++r) { mrow[r] = -1e30f; lrow[r] = 0.f; }

  const int nkb = qb / 32 + 2;          // causal: keys up to qb+63
  for (int kb32 = 0; kb32 < nkb; ++kb32) {
    const int kb = kb32 * 32;

    // ---- batch all global loads first (K and V rows), then cvt+store ----
    v4f kf[2][2], vf[2][2];
    #pragma unroll
    for (int rr = 0; rr < 2; ++rr) {
      const size_t base = ((size_t)b * 512 + kb + skr2 + rr) * rstride + h * 64 + sd8;
      kf[rr][0] = *(const v4f*)(qkv + base + 512);
      kf[rr][1] = *(const v4f*)(qkv + base + 516);
      vf[rr][0] = *(const v4f*)(qkv + base + 1024);
      vf[rr][1] = *(const v4f*)(qkv + base + 1028);
    }
    if (kb32 + 1 < nkb) {
      const size_t nbase = ((size_t)b * 512 + kb + 32 + skr2) * rstride + h * 64 + sd8;
      __builtin_prefetch(qkv + nbase + 512, 0, 3);
      __builtin_prefetch(qkv + nbase + 1024, 0, 3);
    }
    #pragma unroll
    for (int rr = 0; rr < 2; ++rr)
      *(v8bf*)&Ks[skr2 + rr][sd8] = cvt8(kf[rr][0], kf[rr][1]);
    #pragma unroll
    for (int d = 0; d < 8; ++d)           // transpose: (k,k+1) pair -> b32
      *(v2bf*)&Vs[sd8 + d][skr2] =
          cvt2(vf[0][d >> 2][d & 3], vf[1][d >> 2][d & 3]);
    __syncthreads();

    if (kb <= qr0 + 15) {   // wave-uniform: skip fully masked key blocks
      // ---- S = Q K^T * 1/sqrt(dh)  (16 x 32 as two 16x16 D tiles) ----
      v8f sfrag[2];
      #pragma unroll
      for (int nh = 0; nh < 2; ++nh) {
        v8f accs = zero8;
        #pragma unroll
        for (int s = 0; s < 2; ++s) {
          const int kcol = nh * 16 + l15;
          const int dd   = s * 32 + (half16 ? 8 : 0);
          v8bf blo = *(const v8bf*)&Ks[kcol][dd];
          v8bf bhi = *(const v8bf*)&Ks[kcol][dd + 16];
          accs = WMMA_BF16(qfrag[s], pack16(blo, bhi), accs);
        }
        sfrag[nh] = accs * 0.125f;   // 1/sqrt(64)
      }
      // ---- causal mask ----
      #pragma unroll
      for (int nh = 0; nh < 2; ++nh)
        #pragma unroll
        for (int r = 0; r < 8; ++r) {
          const int q_idx = qr0 + r + half16 * 8;
          const int k_idx = kb + nh * 16 + l15;
          sfrag[nh][r] = (k_idx > q_idx) ? -1e30f : sfrag[nh][r];
        }
      // ---- online softmax (rows live in 16-lane halves) ----
      #pragma unroll
      for (int r = 0; r < 8; ++r) {
        float v = fmaxf(sfrag[0][r], sfrag[1][r]);
        #pragma unroll
        for (int m = 1; m <= 8; m <<= 1) v = fmaxf(v, __shfl_xor(v, m, 32));
        const float mnew  = fmaxf(mrow[r], v);
        const float alpha = __expf(mrow[r] - mnew);
        mrow[r] = mnew;
        const float p0 = __expf(sfrag[0][r] - mnew);
        const float p1 = __expf(sfrag[1][r] - mnew);
        sfrag[0][r] = p0; sfrag[1][r] = p1;
        float rs = p0 + p1;
        #pragma unroll
        for (int m = 1; m <= 8; m <<= 1) rs += __shfl_xor(rs, m, 32);
        lrow[r] = lrow[r] * alpha + rs;
        #pragma unroll
        for (int dn = 0; dn < 4; ++dn) o[dn][r] *= alpha;
      }
      // ---- P: D-layout -> A-layout via per-wave LDS region ----
      #pragma unroll
      for (int r = 0; r < 8; ++r) {
        const int prow = r + half16 * 8;
        Ps[wave][prow][l15]      = (bf16_t)sfrag[0][r];
        Ps[wave][prow][16 + l15] = (bf16_t)sfrag[1][r];
      }
      asm volatile("s_wait_dscnt 0" ::: "memory");   // intra-wave LDS RAW
      const int pc0 = half16 ? 8 : 0;
      v8bf plo = *(const v8bf*)&Ps[wave][l15][pc0];
      v8bf phi = *(const v8bf*)&Ps[wave][l15][pc0 + 16];
      v16bf pa = pack16(plo, phi);
      // ---- O += P * V ----
      #pragma unroll
      for (int dn = 0; dn < 4; ++dn) {
        const int dcol = dn * 16 + l15;
        const int kk   = half16 ? 8 : 0;
        v8bf vlo = *(const v8bf*)&Vs[dcol][kk];
        v8bf vhi = *(const v8bf*)&Vs[dcol][kk + 16];
        o[dn] = WMMA_BF16(pa, pack16(vlo, vhi), o[dn]);
      }
    }
    __syncthreads();
  }

  // ---- epilogue: O /= l, write [b][q][h*64+d] f32 ----
  #pragma unroll
  for (int r = 0; r < 8; ++r) lrow[r] = 1.0f / lrow[r];
  #pragma unroll
  for (int dn = 0; dn < 4; ++dn)
    #pragma unroll
    for (int r = 0; r < 8; ++r) {
      const int row = qr0 + r + half16 * 8;
      const int col = h * 64 + dn * 16 + l15;
      out[((size_t)b * 512 + row) * 512 + col] = o[dn][r] * lrow[r];
    }
}

// ---------------------------------------------------------------------------
// dst[i] += src[i], vectorized (for decomposed += prev_decomposed)
// ---------------------------------------------------------------------------
__global__ __launch_bounds__(256)
void add_inplace_kernel(float* __restrict__ dst, const float* __restrict__ src, int n4)
{
  const int i = blockIdx.x * blockDim.x + threadIdx.x;
  if (i < n4) {
    v4f a = *(const v4f*)(dst + (size_t)i * 4);
    v4f b = *(const v4f*)(src + (size_t)i * 4);
    a += b;
    *(v4f*)(dst + (size_t)i * 4) = a;
  }
}

// ---------------------------------------------------------------------------
extern "C" void kernel_launch(void* const* d_in, const int* in_sizes, int n_in,
                              void* d_out, int out_size, void* d_ws, size_t ws_size,
                              hipStream_t stream)
{
  (void)in_sizes; (void)n_in; (void)out_size; (void)ws_size;
  const float* x      = (const float*)d_in[0];
  const float* prev   = (const float*)d_in[1];
  const float* Wattn  = (const float*)d_in[2];
  const float* Wctx   = (const float*)d_in[3];
  const float* Wproj  = (const float*)d_in[4];
  const float* Wproj2 = (const float*)d_in[5];

  float* out_final = (float*)d_out;                       // [64,512,512]
  float* dec_final = out_final + (size_t)64 * 512 * 512;  // [64,512,512]

  float* qkv      = (float*)d_ws;                         // 64*512*1536
  float* attn_out = qkv + (size_t)64 * 512 * 1536;        // 64*512*512
  float* dec      = attn_out + (size_t)64 * 512 * 512;    // 64*512*512

  // 1) qkv = x @ W_attn            (32768 x 1536 x 512)
  gemm_bf16_wmma<<<dim3(1536 / 128, 32768 / 128), 256, 0, stream>>>(
      x, Wattn, qkv, 32768, 1536, 512, 512, 1536, 1536);

  // 2) causal MHA -> attn_out      (512 heads, flash style)
  attention_kernel<<<dim3(8, 8, 64), 128, 0, stream>>>(qkv, attn_out);

  // 3) dec = out[:, -1, :] @ W_ctx (64 x 262144 x 512) — BW-bound, 512MB stream
  gemm_bf16_wmma<<<dim3(262144 / 128, 1), 256, 0, stream>>>(
      attn_out + (size_t)511 * 512, Wctx, dec, 64, 262144, 512,
      512 * 512, 262144, 262144);

  // 4) dec += prev_decomposed
  add_inplace_kernel<<<dim3(16384), 256, 0, stream>>>(dec, prev, 4194304);

  // 5) dec_final = dec @ W_proj2   (32768 x 512 x 512)
  gemm_bf16_wmma<<<dim3(512 / 128, 32768 / 128), 256, 0, stream>>>(
      dec, Wproj2, dec_final, 32768, 512, 512, 512, 512, 512);

  // 6) out_final = attn_out @ W_proj (32768 x 512 x 512)
  gemm_bf16_wmma<<<dim3(512 / 128, 32768 / 128), 256, 0, stream>>>(
      attn_out, Wproj, out_final, 32768, 512, 512, 512, 512, 512);
}
